// EdgeConv_54013508714652
// MI455X (gfx1250) — matrix operand order repaired and checked by reading, not compiled, not run on hardware
//
#include <hip/hip_runtime.h>
#include <hip/hip_bf16.h>
#include <math.h>

// ---------------------------------------------------------------------------
// EdgeConv (DGCNN) for MI455X / gfx1250, wave32, WMMA bf16 split-precision.
//
// Shapes: x (16,64,2048) f32, W (64,128) f32, gamma/beta (64), k=20.
// out (16,64,2048) f32.
//
// Pipeline:
//   k_prep_x  : split x into bf16 hi/lo, packed as (c,c+1) u32 pairs in
//               [b][c2][n] layout (matches WMMA A/B fragment K-pairs);
//               also xx[b][n] = sum_c x^2.
//   k_prep_w  : same bf16 hi/lo split+pack for W, [o][c2] layout.
//   k_dist_topk: per 16-row block, Gram tile via v_wmma_f32_16x16x32_bf16
//               (hi*hi + hi*lo + lo*hi ~ fp32), rank value 2*dot - xx[m],
//               chunked 512-col LDS tile + iterative wave-argmax top-20.
//   k_pq_gemm : P = Xt@W1^T, Q = Xt@(W2-W1)^T via same WMMA path.
//   k_stats   : sum / sumsq of y = P[idx]+Q over all (b,n,k) per channel.
//   k_bnfin   : fold mean/var/gamma/beta into scale/shift.
//   k_out     : y -> BN -> leaky(0.2) -> max over k -> (B,OUT,N) coalesced.
// ---------------------------------------------------------------------------

#define BB   16
#define CD   64
#define NP   2048
#define OUTC 64
#define KNN  20
#define CHUNK 512
#define TPAD  516          // 512 + 4: kills LDS bank conflicts (8*516 % 64 = 32)
#define NCAND 80           // 4 chunks * 20 candidates

typedef __attribute__((ext_vector_type(16))) __bf16 v16bf;
typedef __attribute__((ext_vector_type(8)))  float  v8f;

__device__ __forceinline__ v8f wmma_bf16(v16bf a, v16bf b, v8f c) {
  // D = A(16x32 bf16) * B(32x16 bf16) + C(16x16 f32)
  return __builtin_amdgcn_wmma_f32_16x16x32_bf16(
      /*neg_a=*/false, a, /*neg_b=*/false, b,
      /*c_mod=*/(short)0, c, /*reuse_a=*/false, /*reuse_b=*/false);
}

__device__ __forceinline__ unsigned short f2bf(float f) {
  union { float f; unsigned u; } a; a.f = f;
  unsigned r = a.u + 0x7fffu + ((a.u >> 16) & 1u);   // RNE
  return (unsigned short)(r >> 16);
}
__device__ __forceinline__ float bf2f(unsigned short h) {
  union { unsigned u; float f; } a; a.u = ((unsigned)h) << 16; return a.f;
}

// ---------------------------------------------------------------------------
// Prep x: bf16 hi/lo split, packed K-pairs, plus xx = sum_c x^2.
// xpk layout: u32 at (b*32 + c2)*NP + n holds bf16(x[b][2c2][n]) | bf16(x[b][2c2+1][n])<<16
// ---------------------------------------------------------------------------
__global__ __launch_bounds__(256) void k_prep_x(
    const float* __restrict__ x, unsigned* __restrict__ xpk_hi,
    unsigned* __restrict__ xpk_lo, float* __restrict__ xx) {
  int tid = blockIdx.x * 256 + threadIdx.x;       // 0 .. B*N-1
  int b = tid >> 11, n = tid & (NP - 1);
  float acc = 0.f;
  for (int c2 = 0; c2 < CD / 2; ++c2) {
    float f0 = x[((size_t)(b * CD + 2 * c2)) * NP + n];
    float f1 = x[((size_t)(b * CD + 2 * c2 + 1)) * NP + n];
    unsigned short h0 = f2bf(f0), h1 = f2bf(f1);
    unsigned short l0 = f2bf(f0 - bf2f(h0)), l1 = f2bf(f1 - bf2f(h1));
    size_t off = ((size_t)(b * 32 + c2)) * NP + n;
    xpk_hi[off] = (unsigned)h0 | ((unsigned)h1 << 16);
    xpk_lo[off] = (unsigned)l0 | ((unsigned)l1 << 16);
    acc += f0 * f0 + f1 * f1;
  }
  xx[tid] = acc;
}

// Prep W: wpk[o*64 + c2] packs bf16(W[o][2c2]), bf16(W[o][2c2+1]).
__global__ __launch_bounds__(256) void k_prep_w(
    const float* __restrict__ W, unsigned* __restrict__ wpk_hi,
    unsigned* __restrict__ wpk_lo) {
  for (int e = threadIdx.x; e < OUTC * 64; e += 256) {
    int o = e >> 6, c2 = e & 63;
    float f0 = W[o * 128 + 2 * c2], f1 = W[o * 128 + 2 * c2 + 1];
    unsigned short h0 = f2bf(f0), h1 = f2bf(f1);
    unsigned short l0 = f2bf(f0 - bf2f(h0)), l1 = f2bf(f1 - bf2f(h1));
    wpk_hi[e] = (unsigned)h0 | ((unsigned)h1 << 16);
    wpk_lo[e] = (unsigned)l0 | ((unsigned)l1 << 16);
  }
}

// ---------------------------------------------------------------------------
// Fragment loaders (contiguous K order, same for A and B; wave32).
// A 16x32 bf16: lane l (m=l&15), vgpr v: K = 2*v + (l>=16?16:0)
// B 32x16 bf16: lane l (col=l&15), vgpr v: K = 2*v + (l>=16?16:0)
// K-pairs are even-aligned -> one packed u32 per VGPR.
// ---------------------------------------------------------------------------
__device__ __forceinline__ v16bf load_fragA(const unsigned* __restrict__ xpk,
                                            int b, int n0, int kc,
                                            int jcol, int hi16) {
  union { unsigned u[8]; v16bf v; } r;
#pragma unroll
  for (int v = 0; v < 8; ++v) {
    int k0 = (v << 1) + (hi16 ? 16 : 0);
    int c2 = (kc << 4) + (k0 >> 1);
    r.u[v] = xpk[((size_t)(b * 32 + c2)) * NP + n0 + jcol];
  }
  return r.v;
}
__device__ __forceinline__ v16bf load_fragB_x(const unsigned* __restrict__ xpk,
                                              int b, int m0, int kc,
                                              int jcol, int hi16) {
  union { unsigned u[8]; v16bf v; } r;
#pragma unroll
  for (int v = 0; v < 8; ++v) {
    int k0 = (v << 1) + (hi16 ? 16 : 0);
    int c2 = (kc << 4) + (k0 >> 1);
    r.u[v] = xpk[((size_t)(b * 32 + c2)) * NP + m0 + jcol];
  }
  return r.v;
}
__device__ __forceinline__ v16bf load_fragB_w(const unsigned* __restrict__ wpk,
                                              int o0, int c2base, int kc,
                                              int jcol, int hi16) {
  union { unsigned u[8]; v16bf v; } r;
#pragma unroll
  for (int v = 0; v < 8; ++v) {
    int k0 = (v << 1) + (hi16 ? 16 : 0);
    int c2 = c2base + (kc << 4) + (k0 >> 1);
    r.u[v] = wpk[(o0 + jcol) * 64 + c2];
  }
  return r.v;
}

// split-precision accumulate over full K=64 (2 chunks x {lo*hi, hi*lo, hi*hi})
__device__ __forceinline__ v8f gemm64(const v16bf Ah[2], const v16bf Al[2],
                                      const v16bf Bh[2], const v16bf Bl[2]) {
  v8f acc = {0.f, 0.f, 0.f, 0.f, 0.f, 0.f, 0.f, 0.f};
  acc = wmma_bf16(Al[0], Bh[0], acc);
  acc = wmma_bf16(Ah[0], Bl[0], acc);
  acc = wmma_bf16(Ah[0], Bh[0], acc);
  acc = wmma_bf16(Al[1], Bh[1], acc);
  acc = wmma_bf16(Ah[1], Bl[1], acc);
  acc = wmma_bf16(Ah[1], Bh[1], acc);
  return acc;
}

// ---------------------------------------------------------------------------
// Fused Gram + top-20. One block = (batch, 16 rows). 8 waves.
// Rank value: 2*dot(n,m) - xx[m]   (-xx[n] is row-constant, drop it).
// ---------------------------------------------------------------------------
__global__ __launch_bounds__(256) void k_dist_topk(
    const unsigned* __restrict__ xpk_hi, const unsigned* __restrict__ xpk_lo,
    const float* __restrict__ xx, int* __restrict__ knn_idx) {
  __shared__ float tile[16 * TPAD];
  __shared__ float candV[16 * NCAND];
  __shared__ int   candI[16 * NCAND];

  const int b = blockIdx.x >> 7;
  const int n0 = (blockIdx.x & 127) << 4;
  const int lane = threadIdx.x & 31;
  const int wv = threadIdx.x >> 5;
  const int jcol = lane & 15;
  const int hi16 = lane >> 4;

  v16bf Ah[2], Al[2];
#pragma unroll
  for (int kc = 0; kc < 2; ++kc) {
    Ah[kc] = load_fragA(xpk_hi, b, n0, kc, jcol, hi16);
    Al[kc] = load_fragA(xpk_lo, b, n0, kc, jcol, hi16);
  }

  for (int chunk = 0; chunk < 4; ++chunk) {
    const int cbase = chunk * CHUNK;
    // 32 column tiles per chunk, 4 per wave
    for (int q = 0; q < 4; ++q) {
      const int m0 = cbase + ((wv + (q << 3)) << 4);
      v16bf Bh[2], Bl[2];
#pragma unroll
      for (int kc = 0; kc < 2; ++kc) {
        Bh[kc] = load_fragB_x(xpk_hi, b, m0, kc, jcol, hi16);
        Bl[kc] = load_fragB_x(xpk_lo, b, m0, kc, jcol, hi16);
      }
      v8f acc = gemm64(Ah, Al, Bh, Bl);
      const float xxm = xx[b * NP + m0 + jcol];
      const int colL = (m0 - cbase) + jcol;
#pragma unroll
      for (int r = 0; r < 8; ++r) {
        int row = r + (hi16 << 3);
        tile[row * TPAD + colL] = 2.0f * acc[r] - xxm;
      }
    }
    __syncthreads();

    // this wave extracts top-20 of this chunk for its two rows
    for (int rr = 0; rr < 2; ++rr) {
      const int row = (wv << 1) + rr;
      float* trow = &tile[row * TPAD];
      for (int it = 0; it < KNN; ++it) {
        float bv = -__builtin_huge_valf(); int bi = 0x7fffffff;
#pragma unroll
        for (int j = 0; j < CHUNK / 32; ++j) {
          int c = lane + (j << 5);
          float v = trow[c];
          if (v > bv) { bv = v; bi = c; }       // ascending scan -> low idx wins ties
        }
#pragma unroll
        for (int off = 16; off > 0; off >>= 1) {
          float ov = __shfl_xor(bv, off, 32);
          int   oi = __shfl_xor(bi, off, 32);
          if (ov > bv || (ov == bv && oi < bi)) { bv = ov; bi = oi; }
        }
        if (lane == 0) {
          candV[row * NCAND + chunk * KNN + it] = bv;
          candI[row * NCAND + chunk * KNN + it] = cbase + bi;
          trow[bi] = -__builtin_huge_valf();
        }
        __builtin_amdgcn_wave_barrier();        // keep LDS RAW ordered in-wave
      }
    }
    __syncthreads();
  }

  // merge: exact top-20 of the 80 per-chunk candidates (chunks disjoint)
  for (int rr = 0; rr < 2; ++rr) {
    const int row = (wv << 1) + rr;
    const int n = n0 + row;
    for (int it = 0; it < KNN; ++it) {
      float bv = -__builtin_huge_valf(); int bi = 0x7fffffff, bs = 0;
#pragma unroll
      for (int j = 0; j < 3; ++j) {
        int c = lane + (j << 5);
        if (c < NCAND) {
          float v = candV[row * NCAND + c];
          int  ix = candI[row * NCAND + c];
          if (v > bv || (v == bv && ix < bi)) { bv = v; bi = ix; bs = c; }
        }
      }
#pragma unroll
      for (int off = 16; off > 0; off >>= 1) {
        float ov = __shfl_xor(bv, off, 32);
        int   oi = __shfl_xor(bi, off, 32);
        int   os = __shfl_xor(bs, off, 32);
        if (ov > bv || (ov == bv && oi < bi)) { bv = ov; bi = oi; bs = os; }
      }
      if (lane == 0) {
        knn_idx[(((size_t)b * NP) + n) * KNN + it] = bi;
        candV[row * NCAND + bs] = -__builtin_huge_valf();
      }
      __builtin_amdgcn_wave_barrier();
    }
  }
}

// ---------------------------------------------------------------------------
// P = Xt @ W1^T, Q = Xt @ (W2 - W1)^T.  One wave per 16-row tile, 8 waves/block.
// ---------------------------------------------------------------------------
__global__ __launch_bounds__(256) void k_pq_gemm(
    const unsigned* __restrict__ xpk_hi, const unsigned* __restrict__ xpk_lo,
    const unsigned* __restrict__ wpk_hi, const unsigned* __restrict__ wpk_lo,
    float* __restrict__ P, float* __restrict__ Q) {
  const int wv = threadIdx.x >> 5;
  const int lane = threadIdx.x & 31;
  const int rb = blockIdx.x * 8 + wv;           // 0..2047 row-blocks
  const int b = rb >> 7;
  const int n0 = (rb & 127) << 4;
  const int jcol = lane & 15;
  const int hi16 = lane >> 4;

  v16bf Ah[2], Al[2];
#pragma unroll
  for (int kc = 0; kc < 2; ++kc) {
    Ah[kc] = load_fragA(xpk_hi, b, n0, kc, jcol, hi16);
    Al[kc] = load_fragA(xpk_lo, b, n0, kc, jcol, hi16);
  }

  for (int ot = 0; ot < 4; ++ot) {
    const int o0 = ot << 4;
    v16bf B1h[2], B1l[2], B2h[2], B2l[2];
#pragma unroll
    for (int kc = 0; kc < 2; ++kc) {
      B1h[kc] = load_fragB_w(wpk_hi, o0, 0,  kc, jcol, hi16);   // W1: c2 0..31
      B1l[kc] = load_fragB_w(wpk_lo, o0, 0,  kc, jcol, hi16);
      B2h[kc] = load_fragB_w(wpk_hi, o0, 32, kc, jcol, hi16);   // W2: c2 32..63
      B2l[kc] = load_fragB_w(wpk_lo, o0, 32, kc, jcol, hi16);
    }
    v8f a1 = gemm64(Ah, Al, B1h, B1l);
    v8f a2 = gemm64(Ah, Al, B2h, B2l);
#pragma unroll
    for (int r = 0; r < 8; ++r) {
      int n = n0 + r + (hi16 << 3);
      size_t off = (((size_t)b * NP) + n) * OUTC + o0 + jcol;
      P[off] = a1[r];
      Q[off] = a2[r] - a1[r];                   // Xt @ (W2 - W1)^T
    }
  }
}

__global__ void k_zero(float* __restrict__ sums) { sums[threadIdx.x] = 0.f; }

// ---------------------------------------------------------------------------
// Pass 1: per-channel sum / sumsq of y = P[b,idx,o] + Q[b,n,o].
// block = 256 (4 groups x 64 channels), 32 (b,n) per block.
// ---------------------------------------------------------------------------
__global__ __launch_bounds__(256) void k_stats(
    const float* __restrict__ P, const float* __restrict__ Q,
    const int* __restrict__ knn_idx, float* __restrict__ sums) {
  __shared__ float red[512];
  const int o = threadIdx.x & 63, s = threadIdx.x >> 6;
  float sm = 0.f, ss = 0.f;
  for (int i = 0; i < 8; ++i) {
    int g = blockIdx.x * 32 + s * 8 + i;        // 0..B*N-1
    int b = g >> 11;
    float q = Q[(size_t)g * OUTC + o];
    const int* ip = &knn_idx[(size_t)g * KNN];
#pragma unroll
    for (int k = 0; k < KNN; ++k) {
      float y = P[(((size_t)(b << 11)) + ip[k]) * OUTC + o] + q;
      sm += y; ss += y * y;
    }
  }
  red[threadIdx.x] = sm; red[256 + threadIdx.x] = ss;
  __syncthreads();
  if (threadIdx.x < 64) {
    float s0 = red[o] + red[o + 64] + red[o + 128] + red[o + 192];
    float q0 = red[256 + o] + red[320 + o] + red[384 + o] + red[448 + o];
    atomicAdd(&sums[o], s0);
    atomicAdd(&sums[64 + o], q0);
  }
}

__global__ void k_bnfin(const float* __restrict__ sums,
                        const float* __restrict__ gamma,
                        const float* __restrict__ beta,
                        float* __restrict__ bnp) {
  int t = threadIdx.x;
  const float invc = 1.0f / (float)(BB * NP * KNN);
  float mean = sums[t] * invc;
  float var = sums[64 + t] * invc - mean * mean;
  float inv = rsqrtf(var + 1e-5f);
  float sc = gamma[t] * inv;
  bnp[t] = sc;
  bnp[64 + t] = beta[t] - mean * sc;
}

// ---------------------------------------------------------------------------
// Pass 2: BN + leaky + max over k, coalesced (B,OUT,N) store via LDS transpose.
// block = 256 (4 n-lanes x 64 channels), 64 n per block.
// ---------------------------------------------------------------------------
__global__ __launch_bounds__(256) void k_out(
    const float* __restrict__ P, const float* __restrict__ Q,
    const int* __restrict__ knn_idx, const float* __restrict__ bnp,
    float* __restrict__ out) {
  __shared__ float obuf[64 * 65];
  const int o = threadIdx.x & 63, s = threadIdx.x >> 6;
  const float sc = bnp[o], sh = bnp[64 + o];
  for (int nn = 0; nn < 16; ++nn) {
    int nl = nn * 4 + s;
    int g = blockIdx.x * 64 + nl;
    int b = g >> 11;
    float q = Q[(size_t)g * OUTC + o];
    const int* ip = &knn_idx[(size_t)g * KNN];
    float m = -__builtin_huge_valf();
#pragma unroll
    for (int k = 0; k < KNN; ++k) {
      float y = P[(((size_t)(b << 11)) + ip[k]) * OUTC + o] + q;
      float z = y * sc + sh;
      z = (z >= 0.f) ? z : 0.2f * z;
      m = fmaxf(m, z);
    }
    obuf[o * 65 + nl] = m;
  }
  __syncthreads();
  const int b = (blockIdx.x * 64) >> 11;
  const int nbase = (blockIdx.x * 64) & (NP - 1);
  for (int oo = 0; oo < 16; ++oo) {
    int orow = oo * 4 + s;
    out[(((size_t)b * OUTC) + orow) * NP + nbase + o] = obuf[orow * 65 + o];
  }
}

// ---------------------------------------------------------------------------
extern "C" void kernel_launch(void* const* d_in, const int* in_sizes, int n_in,
                              void* d_out, int out_size, void* d_ws, size_t ws_size,
                              hipStream_t stream) {
  const float* x     = (const float*)d_in[0];   // (16,64,2048)
  const float* W     = (const float*)d_in[1];   // (64,128)
  const float* gamma = (const float*)d_in[2];
  const float* beta  = (const float*)d_in[3];
  float* out = (float*)d_out;

  // workspace layout (~28 MB)
  char* p = (char*)d_ws;
  auto carve = [&](size_t bytes) { char* r = p; p += (bytes + 255) & ~(size_t)255; return r; };
  unsigned* xpk_hi = (unsigned*)carve((size_t)BB * 32 * NP * 4);
  unsigned* xpk_lo = (unsigned*)carve((size_t)BB * 32 * NP * 4);
  unsigned* wpk_hi = (unsigned*)carve(OUTC * 64 * 4);
  unsigned* wpk_lo = (unsigned*)carve(OUTC * 64 * 4);
  float*    xx     = (float*)   carve((size_t)BB * NP * 4);
  int*      knnidx = (int*)     carve((size_t)BB * NP * KNN * 4);
  float*    P      = (float*)   carve((size_t)BB * NP * OUTC * 4);
  float*    Q      = (float*)   carve((size_t)BB * NP * OUTC * 4);
  float*    sums   = (float*)   carve(128 * 4);
  float*    bnp    = (float*)   carve(128 * 4);

  k_prep_x<<<(BB * NP) / 256, 256, 0, stream>>>(x, xpk_hi, xpk_lo, xx);
  k_prep_w<<<1, 256, 0, stream>>>(W, wpk_hi, wpk_lo);
  k_zero<<<1, 128, 0, stream>>>(sums);
  k_dist_topk<<<BB * (NP / 16), 256, 0, stream>>>(xpk_hi, xpk_lo, xx, knnidx);
  k_pq_gemm<<<(BB * (NP / 16)) / 8, 256, 0, stream>>>(xpk_hi, xpk_lo, wpk_hi,
                                                      wpk_lo, P, Q);
  k_stats<<<(BB * NP) / 32, 256, 0, stream>>>(P, Q, knnidx, sums);
  k_bnfin<<<1, 64, 0, stream>>>(sums, gamma, beta, bnp);
  k_out<<<(BB * NP) / 64, 256, 0, stream>>>(P, Q, knnidx, bnp, out);
}
